// SelfAttn_47562467836086
// MI455X (gfx1250) — compile-verified, matrix operand order checked by reading, not run on hardware
//
#include <hip/hip_runtime.h>
#include <hip/hip_bf16.h>
#include <stdint.h>

typedef __attribute__((ext_vector_type(16))) __bf16 v16bf;
typedef __attribute__((ext_vector_type(8)))  float  v8f;
typedef int v4i_vs __attribute__((vector_size(16)));   // pointee type of async-LDS builtin

#define E_DIM 4096
#define NHEAD 32
#define HDIM  128
#define SEQ   2048
#define NB    2
#define BS    (NB * SEQ)   // 4096 rows

// CDNA5 async global->LDS copy path (ASYNCcnt). Guarded so either toolchain
// compiles; fallback is the VGPR round-trip copy.
#if __has_builtin(__builtin_amdgcn_global_load_async_to_lds_b128)
#define HAVE_ASYNC_LDS 1
#else
#define HAVE_ASYNC_LDS 0
#endif

static __device__ __forceinline__ void g2l_async16(unsigned short* lds,
                                                   const unsigned short* g) {
#if HAVE_ASYNC_LDS
    __builtin_amdgcn_global_load_async_to_lds_b128(
        (__attribute__((address_space(1))) v4i_vs*)(uintptr_t)g,
        (__attribute__((address_space(3))) v4i_vs*)(unsigned)(uintptr_t)lds,
        0, 0);
#else
    *(uint4*)lds = *(const uint4*)g;
#endif
}

static __device__ __forceinline__ void wait_async0() {
#if HAVE_ASYNC_LDS
#if __has_builtin(__builtin_amdgcn_s_wait_asynccnt)
    __builtin_amdgcn_s_wait_asynccnt(0);
#else
    asm volatile("s_wait_asynccnt 0" ::: "memory");
#endif
#endif
}

// copy 32 bf16 (4 x b128) global -> LDS, per-thread slice
static __device__ __forceinline__ void stage64(const unsigned short* src,
                                               unsigned short* dst) {
#if HAVE_ASYNC_LDS
#pragma unroll
    for (int i = 0; i < 4; ++i) g2l_async16(dst + i * 8, src + i * 8);
#else
#pragma unroll
    for (int i = 0; i < 4; ++i) ((uint4*)dst)[i] = ((const uint4*)src)[i];
#endif
}

// copy 64 bf16 (8 x b128) global -> LDS, per-thread slice
static __device__ __forceinline__ void stage128(const unsigned short* src,
                                                unsigned short* dst) {
#if HAVE_ASYNC_LDS
#pragma unroll
    for (int i = 0; i < 8; ++i) g2l_async16(dst + i * 8, src + i * 8);
#else
#pragma unroll
    for (int i = 0; i < 8; ++i) ((uint4*)dst)[i] = ((const uint4*)src)[i];
#endif
}

static __device__ __forceinline__ unsigned short f2bf(float f) {
    unsigned u = __float_as_uint(f);
    u += 0x7FFFu + ((u >> 16) & 1u);   // round-to-nearest-even
    return (unsigned short)(u >> 16);
}
static __device__ __forceinline__ float bf2f(unsigned short h) {
    return __uint_as_float(((unsigned)h) << 16);
}

union FragBF {
    v16bf v;
    uint4 q[2];
    unsigned short u[16];
};

// ---------------------------------------------------------------------------
// fp32 -> bf16 converter (grid-stride)
// ---------------------------------------------------------------------------
__global__ void cvt_f32_to_bf16(const float* __restrict__ src,
                                unsigned short* __restrict__ dst, int n) {
    int i = blockIdx.x * blockDim.x + threadIdx.x;
    int stride = gridDim.x * blockDim.x;
    for (; i < n; i += stride) dst[i] = f2bf(src[i]);
}

// ---------------------------------------------------------------------------
// y[M,N] = A[M,K] * Bw[N,K]^T + bias ; bf16 in, fp32 accumulate.
// 128x128 block tile, 8 waves as 4(M) x 2(N), each wave 2x4 16x16 WMMA tiles.
// Double-buffered LDS, async global->LDS staging overlapped with WMMAs.
// out_mode: 0 = bf16 row-major [M,N]
//           1 = fp32 row-major [M,N]
//           2 = bf16 transposed head-major [B, H, D, S] (for attention V)
// ---------------------------------------------------------------------------
__global__ __launch_bounds__(256) void gemm_bf16_nt(
    const unsigned short* __restrict__ A,    // [M,K] bf16
    const unsigned short* __restrict__ Bw,   // [N,K] bf16 (torch Linear weight)
    const float* __restrict__ bias,          // [N] fp32
    void* __restrict__ C, int M, int N, int K, int out_mode)
{
    extern __shared__ unsigned short gsm[];
    unsigned short* As0 = gsm;                 // [128][72]
    unsigned short* Bs0 = gsm + 9216;
    unsigned short* As1 = gsm + 18432;
    unsigned short* Bs1 = gsm + 27648;         // total 73728 B

    const int tid  = threadIdx.x;
    const int wave = tid >> 5, lane = tid & 31;
    const int wm = wave & 3;            // wave row (x32 rows)
    const int wn = wave >> 2;           // wave col (x64 cols)
    const int m_blk = blockIdx.y * 128, n_blk = blockIdx.x * 128;
    const int lrow = lane & 15;
    const int c0   = (lane < 16) ? 0 : 8;   // K-chunk base per ISA 16-bit layout

    v8f zero = {};
    v8f acc[2][4];
#pragma unroll
    for (int i = 0; i < 2; ++i)
#pragma unroll
        for (int j = 0; j < 4; ++j) acc[i][j] = zero;

    const int  ldrow = tid >> 1;          // 0..127
    const int  ldcol = (tid & 1) * 32;    // 0 or 32
    const long arow  = (long)(m_blk + ldrow) * K + ldcol;
    const long brow  = (long)(n_blk + ldrow) * K + ldcol;
    const int  ldoff = ldrow * 72 + ldcol;

    // prologue: stage chunk 0 into buffer 0
    stage64(A + arow, As0 + ldoff);
    stage64(Bw + brow, Bs0 + ldoff);
    wait_async0();
    __syncthreads();

    int buf = 0;
    for (int k0 = 0; k0 < K; k0 += 64) {
        unsigned short* Asb = buf ? As1 : As0;
        unsigned short* Bsb = buf ? Bs1 : Bs0;
        if (k0 + 64 < K) {   // stage next chunk into the other buffer (async)
            unsigned short* Asn = buf ? As0 : As1;
            unsigned short* Bsn = buf ? Bs0 : Bs1;
            stage64(A + arow + k0 + 64, Asn + ldoff);
            stage64(Bw + brow + k0 + 64, Bsn + ldoff);
        }
        if (k0 + 128 < K) {  // prefetch the chunk after that
            __builtin_prefetch(A + arow + k0 + 128, 0, 0);
            __builtin_prefetch(Bw + brow + k0 + 128, 0, 0);
        }

#pragma unroll
        for (int kk = 0; kk < 64; kk += 32) {
            FragBF a[2], b[4];
#pragma unroll
            for (int i = 0; i < 2; ++i) {
                int r = wm * 32 + i * 16 + lrow;
                a[i].q[0] = *(const uint4*)&Asb[r * 72 + kk + c0];
                a[i].q[1] = *(const uint4*)&Asb[r * 72 + kk + c0 + 16];
            }
#pragma unroll
            for (int j = 0; j < 4; ++j) {
                int r = wn * 64 + j * 16 + lrow;
                b[j].q[0] = *(const uint4*)&Bsb[r * 72 + kk + c0];
                b[j].q[1] = *(const uint4*)&Bsb[r * 72 + kk + c0 + 16];
            }
#pragma unroll
            for (int i = 0; i < 2; ++i)
#pragma unroll
                for (int j = 0; j < 4; ++j)
                    acc[i][j] = __builtin_amdgcn_wmma_f32_16x16x32_bf16(
                        false, a[i].v, false, b[j].v, (short)0, acc[i][j], false, false);
        }

        wait_async0();       // this wave's async copies into next buffer landed
        __syncthreads();     // all waves done reading current buffer
        buf ^= 1;
    }

    // epilogue: C-layout lane (m = vj + 8*(lane>=16), n = lane&15)
    const int mo = (lane < 16) ? 0 : 8;
#pragma unroll
    for (int i = 0; i < 2; ++i) {
#pragma unroll
        for (int j = 0; j < 4; ++j) {
            int nn = n_blk + wn * 64 + j * 16 + lrow;
            float bv = bias[nn];
#pragma unroll
            for (int vj = 0; vj < 8; ++vj) {
                int mm = m_blk + wm * 32 + i * 16 + vj + mo;
                float val = acc[i][j][vj] + bv;
                if (out_mode == 1) {
                    ((float*)C)[(long)mm * N + nn] = val;
                } else if (out_mode == 2) {
                    // [B,S,(H*D)] -> [B,H,D,S]: row mm = b*SEQ+s, col nn = h*D+d
                    int bb = mm / SEQ, ss = mm - bb * SEQ;
                    long off = ((long)bb * (NHEAD * HDIM) + nn) * SEQ + ss;
                    ((unsigned short*)C)[off] = f2bf(val);
                } else {
                    ((unsigned short*)C)[(long)mm * N + nn] = f2bf(val);
                }
            }
        }
    }
}

// ---------------------------------------------------------------------------
// RoPE in-place on bf16 [B*S, E]; faithful to reference: dim = n_heads = 32,
// 16 freqs, repeat_interleave x4, fp32 overflow (theta^e -> inf -> freq 0).
// ---------------------------------------------------------------------------
__global__ void rope_bf16(unsigned short* __restrict__ qk, int total_pairs) {
    int i = blockIdx.x * blockDim.x + threadIdx.x;
    int stride = gridDim.x * blockDim.x;
    const int halfE = E_DIM / 2;
    for (; i < total_pairs; i += stride) {
        int m  = i / halfE;
        int pa = i - m * halfE;
        int h  = pa / (HDIM / 2);
        int p  = pa - h * (HDIM / 2);     // pair index 0..63
        int s  = m % SEQ;                 // position
        int j  = p >> 2;                  // freq index 0..15 (rep=4)
        float pw = powf(10000.0f, (float)(2 * j));   // inf for j>=5 in fp32
        float freq = 32.0f / pw;                      // dim / theta^e ; inf -> 0
        float ang = (float)s * freq;
        float c = cosf(ang), sn = sinf(ang);
        long base = (long)m * E_DIM + h * HDIM + 2 * p;
        float x1 = bf2f(qk[base]), x2 = bf2f(qk[base + 1]);
        qk[base]     = f2bf(x1 * c - x2 * sn);
        qk[base + 1] = f2bf(x1 * sn + x2 * c);
    }
}

// ---------------------------------------------------------------------------
// Flash attention (non-causal). Grid (S/128, H, B), 8 waves/block, each wave
// owns 16 query rows x D=128. Double-buffered K/V chunks of 128 keys staged
// in LDS via async copies, overlapped with WMMA compute.
// Q,K: [B*S, E] bf16 (row-major). Vt: [B, H, D, S] bf16 (key-contiguous).
// ---------------------------------------------------------------------------
__global__ __launch_bounds__(256) void flash_attn_bf16(
    const unsigned short* __restrict__ Q,
    const unsigned short* __restrict__ K,
    const unsigned short* __restrict__ Vt,
    unsigned short* __restrict__ O)         // [B*S, E] bf16
{
    extern __shared__ unsigned short smem[];
    unsigned short* Ks0 = smem;                      // [128 keys][136]
    unsigned short* Vs0 = smem + 17408;              // [128 d   ][136]
    unsigned short* Ks1 = smem + 34816;
    unsigned short* Vs1 = smem + 52224;
    unsigned short* Ps  = smem + 69632;              // 8 waves x [16][136]

    const int tid  = threadIdx.x;
    const int wave = tid >> 5, lane = tid & 31;
    const int lrow = lane & 15;
    const int c0   = (lane < 16) ? 0 : 8;
    const int b = blockIdx.z, h = blockIdx.y;
    const int q0 = blockIdx.x * 128;

    unsigned short* Pw = Ps + wave * 16 * 136;

    // Q fragments for this wave's 16 rows (A-layout, direct global b128 loads)
    FragBF aq[4];
    {
        long qrow = (long)(b * SEQ + q0 + wave * 16 + lrow) * E_DIM + h * HDIM;
#pragma unroll
        for (int kk = 0; kk < 4; ++kk) {
            aq[kk].q[0] = *(const uint4*)(Q + qrow + kk * 32 + c0);
            aq[kk].q[1] = *(const uint4*)(Q + qrow + kk * 32 + c0 + 16);
        }
    }

    v8f zero = {};
    v8f o_acc[8];
    float mrun[8], lrun[8];
#pragma unroll
    for (int t = 0; t < 8; ++t) o_acc[t] = zero;
#pragma unroll
    for (int j = 0; j < 8; ++j) { mrun[j] = -1e30f; lrun[j] = 0.0f; }

    const float scale = 0.08838834764831845f;   // 1/sqrt(128)

    const int  ldrow = tid >> 1;          // 0..127 (key row for K, d row for Vt)
    const int  ldcol = (tid & 1) * 64;    // 0 or 64
    const int  ldoff = ldrow * 136 + ldcol;
    // K chunk: rows are keys, row stride E_DIM; chunk advances by 128 rows
    const long kbase = (long)(b * SEQ + ldrow) * E_DIM + h * HDIM + ldcol;
    // Vt chunk: rows are d, row stride SEQ; chunk advances by 128 columns (keys)
    const long vbase = ((long)(b * NHEAD + h) * HDIM + ldrow) * SEQ + ldcol;

    // prologue: stage key-chunk 0 into buffer 0
    stage128(K + kbase, Ks0 + ldoff);
    stage128(Vt + vbase, Vs0 + ldoff);
    wait_async0();
    __syncthreads();

    int buf = 0;
    for (int kc = 0; kc < SEQ; kc += 128) {
        unsigned short* Ksb = buf ? Ks1 : Ks0;
        unsigned short* Vsb = buf ? Vs1 : Vs0;
        if (kc + 128 < SEQ) {   // async-stage next chunk into the other buffer
            unsigned short* Ksn = buf ? Ks0 : Ks1;
            unsigned short* Vsn = buf ? Vs0 : Vs1;
            stage128(K + kbase + (long)(kc + 128) * E_DIM, Ksn + ldoff);
            stage128(Vt + vbase + (kc + 128), Vsn + ldoff);
        }

        // S = Q K^T  (8 key subtiles x 4 K-steps = 32 WMMA)
        v8f sc[8];
#pragma unroll
        for (int t = 0; t < 8; ++t) {
            v8f a = zero;
#pragma unroll
            for (int kk = 0; kk < 4; ++kk) {
                FragBF bk;
                int r = t * 16 + lrow;
                bk.q[0] = *(const uint4*)(Ksb + r * 136 + kk * 32 + c0);
                bk.q[1] = *(const uint4*)(Ksb + r * 136 + kk * 32 + c0 + 16);
                a = __builtin_amdgcn_wmma_f32_16x16x32_bf16(
                        false, aq[kk].v, false, bk.v, (short)0, a, false, false);
            }
            sc[t] = a;
        }

        // online softmax (rows of a C-tile live across 16 lanes of a half-wave)
#pragma unroll
        for (int j = 0; j < 8; ++j) {
            float mx = sc[0][j];
#pragma unroll
            for (int t = 1; t < 8; ++t) mx = fmaxf(mx, sc[t][j]);
            mx *= scale;
#pragma unroll
            for (int off = 8; off >= 1; off >>= 1)
                mx = fmaxf(mx, __shfl_xor(mx, off, 32));
            float mnew  = fmaxf(mrun[j], mx);
            float alpha = __expf(mrun[j] - mnew);
            float rsum = 0.0f;
#pragma unroll
            for (int t = 0; t < 8; ++t) {
                float p = __expf(sc[t][j] * scale - mnew);
                sc[t][j] = p;
                rsum += p;
            }
#pragma unroll
            for (int off = 8; off >= 1; off >>= 1)
                rsum += __shfl_xor(rsum, off, 32);
            lrun[j] = lrun[j] * alpha + rsum;
            mrun[j] = mnew;
#pragma unroll
            for (int t = 0; t < 8; ++t) o_acc[t][j] *= alpha;
        }

        // re-layout P: C-layout -> LDS row-major (per-wave scratch, no barrier)
        {
            int mo = (lane < 16) ? 0 : 8;
#pragma unroll
            for (int j = 0; j < 8; ++j)
#pragma unroll
                for (int t = 0; t < 8; ++t)
                    Pw[(j + mo) * 136 + t * 16 + lrow] = f2bf(sc[t][j]);
        }

        // O += P @ V  (V B-fragments now contiguous b128 loads from Vt rows)
        FragBF ap[4];
#pragma unroll
        for (int kk = 0; kk < 4; ++kk) {
            ap[kk].q[0] = *(const uint4*)(Pw + lrow * 136 + kk * 32 + c0);
            ap[kk].q[1] = *(const uint4*)(Pw + lrow * 136 + kk * 32 + c0 + 16);
        }
#pragma unroll
        for (int t = 0; t < 8; ++t) {
            int d = t * 16 + lrow;
#pragma unroll
            for (int kk = 0; kk < 4; ++kk) {
                FragBF bv;
                bv.q[0] = *(const uint4*)(Vsb + d * 136 + kk * 32 + c0);
                bv.q[1] = *(const uint4*)(Vsb + d * 136 + kk * 32 + c0 + 16);
                o_acc[t] = __builtin_amdgcn_wmma_f32_16x16x32_bf16(
                               false, ap[kk].v, false, bv.v, (short)0, o_acc[t], false, false);
            }
        }

        wait_async0();     // next chunk landed in LDS
        __syncthreads();   // everyone done reading current chunk
        buf ^= 1;
    }

    // normalize and store bf16
    {
        int mo = (lane < 16) ? 0 : 8;
#pragma unroll
        for (int j = 0; j < 8; ++j) {
            float inv = 1.0f / lrun[j];
            long row = (long)(b * SEQ + q0 + wave * 16 + j + mo);
#pragma unroll
            for (int t = 0; t < 8; ++t) {
                float val = o_acc[t][j] * inv;
                O[row * E_DIM + h * HDIM + t * 16 + lrow] = f2bf(val);
            }
        }
    }
}

// ---------------------------------------------------------------------------
// host launcher
// ---------------------------------------------------------------------------
extern "C" void kernel_launch(void* const* d_in, const int* in_sizes, int n_in,
                              void* d_out, int out_size, void* d_ws, size_t ws_size,
                              hipStream_t stream) {
    const float* x  = (const float*)d_in[0];
    const float* Wq = (const float*)d_in[1];
    const float* bq = (const float*)d_in[2];
    const float* Wk = (const float*)d_in[3];
    const float* bk = (const float*)d_in[4];
    const float* Wv = (const float*)d_in[5];
    const float* bv = (const float*)d_in[6];
    const float* Wo = (const float*)d_in[7];
    const float* bo = (const float*)d_in[8];

    const long NE = (long)BS * E_DIM;       // activation elements
    const long WE = (long)E_DIM * E_DIM;    // weight elements

    char* ws = (char*)d_ws;
    unsigned short* xb  = (unsigned short*)ws; ws += NE * 2;
    unsigned short* wqb = (unsigned short*)ws; ws += WE * 2;
    unsigned short* wkb = (unsigned short*)ws; ws += WE * 2;
    unsigned short* wvb = (unsigned short*)ws; ws += WE * 2;
    unsigned short* wob = (unsigned short*)ws; ws += WE * 2;
    unsigned short* qb  = (unsigned short*)ws; ws += NE * 2;
    unsigned short* kb  = (unsigned short*)ws; ws += NE * 2;
    unsigned short* vtb = (unsigned short*)ws; ws += NE * 2;   // [B,H,D,S]
    unsigned short* ab  = (unsigned short*)ws; ws += NE * 2;

    // 1) convert to bf16
    cvt_f32_to_bf16<<<2048, 256, 0, stream>>>(x,  xb,  (int)NE);
    cvt_f32_to_bf16<<<2048, 256, 0, stream>>>(Wq, wqb, (int)WE);
    cvt_f32_to_bf16<<<2048, 256, 0, stream>>>(Wk, wkb, (int)WE);
    cvt_f32_to_bf16<<<2048, 256, 0, stream>>>(Wv, wvb, (int)WE);
    cvt_f32_to_bf16<<<2048, 256, 0, stream>>>(Wo, wob, (int)WE);

    // 2) QKV projections (double-buffered LDS: 73,728 B dynamic)
    //    V is written directly in transposed [B,H,D,S] layout (out_mode 2).
    dim3 ggrid(E_DIM / 128, BS / 128);   // (32, 32)
    size_t gemm_lds = 4u * 9216u * sizeof(unsigned short);
    gemm_bf16_nt<<<ggrid, 256, gemm_lds, stream>>>(xb, wqb, bq, qb,  BS, E_DIM, E_DIM, 0);
    gemm_bf16_nt<<<ggrid, 256, gemm_lds, stream>>>(xb, wkb, bk, kb,  BS, E_DIM, E_DIM, 0);
    gemm_bf16_nt<<<ggrid, 256, gemm_lds, stream>>>(xb, wvb, bv, vtb, BS, E_DIM, E_DIM, 2);

    // 3) RoPE on Q and K (pairs per tensor = BS * E/2)
    int pairs = (int)(NE / 2);
    rope_bf16<<<2048, 256, 0, stream>>>(qb, pairs);
    rope_bf16<<<2048, 256, 0, stream>>>(kb, pairs);

    // 4) fused attention (double-buffered K/Vt + P scratch: 174,080 B dynamic)
    size_t attn_lds = (size_t)(69632 + 17408) * sizeof(unsigned short);
    flash_attn_bf16<<<dim3(SEQ / 128, NHEAD, NB), 256, attn_lds, stream>>>(qb, kb, vtb, ab);

    // 5) output projection -> fp32 d_out
    gemm_bf16_nt<<<ggrid, 256, gemm_lds, stream>>>(ab, wob, bo, d_out, BS, E_DIM, E_DIM, 1);
}
